// WindowAttentionLayer_7550552506691
// MI455X (gfx1250) — matrix operand order repaired and checked by reading, not compiled, not run on hardware
//
#include <hip/hip_runtime.h>

typedef __bf16 bf16;
typedef __attribute__((ext_vector_type(16))) __bf16 bf16x16;
typedef __attribute__((ext_vector_type(8)))  __bf16 bf16x8;
typedef __attribute__((ext_vector_type(8)))  float  f32x8;

static __device__ __forceinline__ bf16x16 catv(bf16x8 lo, bf16x8 hi) {
    bf16x16 r;
#pragma unroll
    for (int i = 0; i < 8; ++i) { r[i] = lo[i]; r[i + 8] = hi[i]; }
    return r;
}

static __device__ __forceinline__ bf16x8 ld8g(const bf16* p) {
    return *reinterpret_cast<const bf16x8*>(p);
}

// A-matrix fragment (16x32 bf16, M x K), row-major source, contiguous along K.
// lane<16: elems 0..7 = K[k0+0..7],  8..15 = K[k0+16..23]
// lane>=16: elems 0..7 = K[k0+8..15], 8..15 = K[k0+24..31]
static __device__ __forceinline__ bf16x16 load_a(const bf16* __restrict__ base, int ld,
                                                 int row, int k0, int lane) {
    const int sel8 = (lane & 16) ? 8 : 0;
    const bf16* p = base + (size_t)row * ld + k0 + sel8;
    return catv(ld8g(p), ld8g(p + 16));
}

// B-matrix fragment (32x16 bf16, K x N) where column n of B is row (n) of the
// row-major source (NT layout): lane holds column lane%16, same K order as load_a.
static __device__ __forceinline__ bf16x16 load_b(const bf16* __restrict__ base, int ld,
                                                 int col_row, int k0, int lane) {
    const int sel16 = (lane & 16) ? 8 : 0;
    const bf16* p = base + (size_t)col_row * ld + k0 + sel16;
    return catv(ld8g(p), ld8g(p + 16));
}

static __device__ __forceinline__ f32x8 wmma_bf16(bf16x16 a, bf16x16 b, f32x8 c) {
    return __builtin_amdgcn_wmma_f32_16x16x32_bf16(false, a, false, b, (short)0, c,
                                                   false, false);
}

static __device__ __forceinline__ f32x8 zero8() {
    f32x8 z;
#pragma unroll
    for (int i = 0; i < 8; ++i) z[i] = 0.0f;
    return z;
}

// ---------------- f32 -> bf16 conversion ----------------
__global__ void cvt_bf16_kernel(const float* __restrict__ src, bf16* __restrict__ dst, int n) {
    int i = blockIdx.x * 256 + threadIdx.x;
    if (i < n) dst[i] = (bf16)src[i];
}

// ---------------- QKV GEMM: [1024,1024] x [3072,1024]^T ----------------
// wave computes 16x64; outputs routed to q(scaled)/k ([H][N][64]) and vT ([H][64][N])
__global__ __launch_bounds__(128) void qkv_gemm_kernel(
    const bf16* __restrict__ xb, const bf16* __restrict__ wb,
    const float* __restrict__ bqkv,
    bf16* __restrict__ qb, bf16* __restrict__ kb, bf16* __restrict__ vT) {
    const int lane = threadIdx.x & 31;
    const int wid  = threadIdx.x >> 5;
    const int m0 = blockIdx.x * 16;
    const int n0 = (blockIdx.y * 4 + wid) * 64;
    const int nI = lane & 15, half = lane >> 4;

    f32x8 c[4];
#pragma unroll
    for (int t = 0; t < 4; ++t) c[t] = zero8();

    const int rowA = m0 + nI;
    const int rowB = n0 + nI;
    for (int k0 = 0; k0 < 1024; k0 += 32) {
        bf16x16 a = load_a(xb, 1024, rowA, k0, lane);
#pragma unroll
        for (int t = 0; t < 4; ++t) {
            bf16x16 b = load_b(wb, 1024, rowB + 16 * t, k0, lane);
            c[t] = wmma_bf16(a, b, c[t]);
        }
    }

    const int s  = n0 >> 10;        // 0=q 1=k 2=v
    const int hH = (n0 >> 6) & 15;  // head
#pragma unroll
    for (int t = 0; t < 4; ++t) {
        const int colbase = n0 + 16 * t;
        const int d = (colbase & 63) + nI;   // dh index
        const float bias = bqkv[colbase + nI];
        if (s == 2) {
            bf16x8 pk;
#pragma unroll
            for (int r = 0; r < 8; ++r) pk[r] = (bf16)(c[t][r] + bias);
            const int nbase = m0 + 8 * half;
            *reinterpret_cast<bf16x8*>(vT + ((size_t)hH * 64 + d) * 1024 + nbase) = pk;
        } else {
            bf16* dst = (s == 0) ? qb : kb;
            const float sc = (s == 0) ? 0.125f : 1.0f;  // dh^-0.5 = 1/8
#pragma unroll
            for (int r = 0; r < 8; ++r) {
                const int n = m0 + r + 8 * half;
                dst[((size_t)hH * 1024 + n) * 64 + d] = (bf16)((c[t][r] + bias) * sc);
            }
        }
    }
}

// ---------------- Flash attention ----------------
// one wave per (head, 16-query-row tile); key blocks of 32; online softmax.
#define AT_WAVES 4
__global__ __launch_bounds__(128) void attn_kernel(
    const bf16* __restrict__ qb, const bf16* __restrict__ kb,
    const bf16* __restrict__ vT, const float* __restrict__ bias_table,
    bf16* __restrict__ outb) {
    __shared__ bf16  Psh[AT_WAVES][16 * 40];   // P row-major, padded stride 40
    __shared__ float Teff[AT_WAVES][128];      // collapsed rel-pos bias table

    const int lane = threadIdx.x & 31;
    const int wid  = threadIdx.x >> 5;
    const int gid  = blockIdx.x * AT_WAVES + wid;
    const int h    = gid >> 6;
    const int q0   = (gid & 63) * 16;
    const int nI   = lane & 15, half = lane >> 4;

    // bias index depends only on s = (yq+xq) - (yk+xk) + 62 in [0,124];
    // raw table index 63*s is clamped JAX-style to 3968.
    for (int ss = lane; ss < 125; ss += 32) {
        int idx = 63 * ss;
        if (idx > 3968) idx = 3968;
        Teff[wid][ss] = bias_table[(size_t)idx * 16 + h];
    }

    const bf16* qh = qb + (size_t)h * 1024 * 64;
    const bf16* kh = kb + (size_t)h * 1024 * 64;
    const bf16* vh = vT + (size_t)h * 64 * 1024;

    const bf16x16 aq0 = load_a(qh, 64, q0 + nI, 0, lane);
    const bf16x16 aq1 = load_a(qh, 64, q0 + nI, 32, lane);

    f32x8 acc[4];
#pragma unroll
    for (int t = 0; t < 4; ++t) acc[t] = zero8();
    float mrow[8], lrow[8];
    int fq[8];
#pragma unroll
    for (int r = 0; r < 8; ++r) {
        mrow[r] = -3.0e38f;
        lrow[r] = 0.0f;
        const int n = q0 + r + 8 * half;
        fq[r] = (n >> 5) + (n & 31);
    }
    __asm__ volatile("s_wait_dscnt 0" ::: "memory");  // Teff visible to whole wave

    for (int kb2 = 0; kb2 < 1024; kb2 += 32) {
        // ---- S = q @ k^T for 32 keys (two 16x16 tiles) ----
        f32x8 s0 = zero8(), s1 = zero8();
        {
            bf16x16 b00 = load_b(kh, 64, kb2 + nI, 0, lane);
            bf16x16 b01 = load_b(kh, 64, kb2 + nI, 32, lane);
            s0 = wmma_bf16(aq0, b00, s0);
            s0 = wmma_bf16(aq1, b01, s0);
            bf16x16 b10 = load_b(kh, 64, kb2 + 16 + nI, 0, lane);
            bf16x16 b11 = load_b(kh, 64, kb2 + 16 + nI, 32, lane);
            s1 = wmma_bf16(aq0, b10, s1);
            s1 = wmma_bf16(aq1, b11, s1);
        }
        // ---- relative position bias ----
        const int m0k = kb2 + nI, m1k = m0k + 16;
        const int fk0 = (m0k >> 5) + (m0k & 31);
        const int fk1 = (m1k >> 5) + (m1k & 31);
#pragma unroll
        for (int r = 0; r < 8; ++r) {
            s0[r] += Teff[wid][fq[r] - fk0 + 62];
            s1[r] += Teff[wid][fq[r] - fk1 + 62];
        }
        // ---- online softmax ----
        float mx[8];
#pragma unroll
        for (int r = 0; r < 8; ++r) mx[r] = fmaxf(s0[r], s1[r]);
#pragma unroll
        for (int r = 0; r < 8; ++r)
#pragma unroll
            for (int off = 1; off < 16; off <<= 1)
                mx[r] = fmaxf(mx[r], __shfl_xor(mx[r], off, 16));
        float alpha[8], rs[8];
#pragma unroll
        for (int r = 0; r < 8; ++r) {
            const float mn = fmaxf(mrow[r], mx[r]);
            alpha[r] = __expf(mrow[r] - mn);
            mrow[r] = mn;
            s0[r] = __expf(s0[r] - mn);
            s1[r] = __expf(s1[r] - mn);
            rs[r] = s0[r] + s1[r];
        }
#pragma unroll
        for (int r = 0; r < 8; ++r)
#pragma unroll
            for (int off = 1; off < 16; off <<= 1)
                rs[r] += __shfl_xor(rs[r], off, 16);
#pragma unroll
        for (int r = 0; r < 8; ++r) lrow[r] = lrow[r] * alpha[r] + rs[r];
#pragma unroll
        for (int t = 0; t < 4; ++t)
#pragma unroll
            for (int r = 0; r < 8; ++r) acc[t][r] *= alpha[r];

        // ---- stage P (16x32 bf16) through LDS to get A-fragment layout ----
        bf16* P = &Psh[wid][0];
#pragma unroll
        for (int r = 0; r < 8; ++r) {
            const int M = r + 8 * half;
            P[M * 40 + nI]      = (bf16)s0[r];
            P[M * 40 + 16 + nI] = (bf16)s1[r];
        }
        __asm__ volatile("s_wait_dscnt 0" ::: "memory");
        const bf16* pr = P + nI * 40 + half * 8;
        bf16x16 aP = catv(*reinterpret_cast<const bf16x8*>(pr),
                          *reinterpret_cast<const bf16x8*>(pr + 16));

        // ---- O += P @ V (v stored transposed: contiguous along keys) ----
#pragma unroll
        for (int t = 0; t < 4; ++t) {
            bf16x16 bv = load_b(vh, 1024, 16 * t + nI, kb2, lane);
            acc[t] = wmma_bf16(aP, bv, acc[t]);
        }
    }

    // ---- normalize and store merged-head output (bf16 [N][D]) ----
    float inv[8];
#pragma unroll
    for (int r = 0; r < 8; ++r) inv[r] = 1.0f / lrow[r];
#pragma unroll
    for (int t = 0; t < 4; ++t)
#pragma unroll
        for (int r = 0; r < 8; ++r) {
            const int n = q0 + r + 8 * half;
            outb[(size_t)n * 1024 + h * 64 + 16 * t + nI] = (bf16)(acc[t][r] * inv[r]);
        }
}

// ---------------- Output projection: y = outb @ w_proj^T + b_proj (f32) ----------------
__global__ __launch_bounds__(128) void proj_gemm_kernel(
    const bf16* __restrict__ ab, const bf16* __restrict__ wb,
    const float* __restrict__ bproj, float* __restrict__ y) {
    const int lane = threadIdx.x & 31;
    const int wid  = threadIdx.x >> 5;
    const int m0 = blockIdx.x * 16;
    const int n0 = (blockIdx.y * 4 + wid) * 64;
    const int nI = lane & 15, half = lane >> 4;

    f32x8 c[4];
#pragma unroll
    for (int t = 0; t < 4; ++t) c[t] = zero8();

    const int rowA = m0 + nI;
    const int rowB = n0 + nI;
    for (int k0 = 0; k0 < 1024; k0 += 32) {
        bf16x16 a = load_a(ab, 1024, rowA, k0, lane);
#pragma unroll
        for (int t = 0; t < 4; ++t) {
            bf16x16 b = load_b(wb, 1024, rowB + 16 * t, k0, lane);
            c[t] = wmma_bf16(a, b, c[t]);
        }
    }
#pragma unroll
    for (int t = 0; t < 4; ++t) {
        const int col = n0 + 16 * t + nI;
        const float bb = bproj[col];
#pragma unroll
        for (int r = 0; r < 8; ++r) {
            const int n = m0 + r + 8 * half;
            y[(size_t)n * 1024 + col] = c[t][r] + bb;
        }
    }
}

extern "C" void kernel_launch(void* const* d_in, const int* in_sizes, int n_in,
                              void* d_out, int out_size, void* d_ws, size_t ws_size,
                              hipStream_t stream) {
    const float* x          = (const float*)d_in[0];   // [1024,1024]
    const float* w_qkv      = (const float*)d_in[1];   // [3072,1024]
    const float* b_qkv      = (const float*)d_in[2];   // [3072]
    const float* w_proj     = (const float*)d_in[3];   // [1024,1024]
    const float* b_proj     = (const float*)d_in[4];   // [1024]
    const float* bias_table = (const float*)d_in[5];   // [3969,16]
    float* y = (float*)d_out;

    const size_t MB = 1u << 20;
    char* ws = (char*)d_ws;
    bf16* xb     = (bf16*)(ws + 0 * MB);   // 2 MB
    bf16* wqkvb  = (bf16*)(ws + 2 * MB);   // 6 MB
    bf16* wprojb = (bf16*)(ws + 8 * MB);   // 2 MB
    bf16* qb     = (bf16*)(ws + 10 * MB);  // 2 MB  [H][N][64]
    bf16* kb     = (bf16*)(ws + 12 * MB);  // 2 MB  [H][N][64]
    bf16* vT     = (bf16*)(ws + 14 * MB);  // 2 MB  [H][64][N]
    bf16* outb   = (bf16*)(ws + 16 * MB);  // 2 MB  [N][D]

    cvt_bf16_kernel<<<(1024 * 1024 + 255) / 256, 256, 0, stream>>>(x, xb, 1024 * 1024);
    cvt_bf16_kernel<<<(3072 * 1024 + 255) / 256, 256, 0, stream>>>(w_qkv, wqkvb, 3072 * 1024);
    cvt_bf16_kernel<<<(1024 * 1024 + 255) / 256, 256, 0, stream>>>(w_proj, wprojb, 1024 * 1024);

    qkv_gemm_kernel<<<dim3(64, 12), 128, 0, stream>>>(xb, wqkvb, b_qkv, qb, kb, vT);
    attn_kernel<<<256, 128, 0, stream>>>(qb, kb, vT, bias_table, outb);
    proj_gemm_kernel<<<dim3(64, 4), 128, 0, stream>>>(outb, wprojb, b_proj, y);
}
